// I_45810121179238
// MI455X (gfx1250) — matrix operand order repaired and checked by reading, not compiled, not run on hardware
//
#include <hip/hip_runtime.h>
#include <hip/hip_bf16.h>

typedef __attribute__((ext_vector_type(16))) _Float16 v16h;
typedef __attribute__((ext_vector_type(8)))  _Float16 v8h;
typedef __attribute__((ext_vector_type(8)))  float    v8f;

#define D_DIM 1024

// ---- monotonic float <-> uint encoding so we can use integer atomicMax ----
__device__ __forceinline__ unsigned enc_f32(float x) {
    unsigned u = __float_as_uint(x);
    return (u & 0x80000000u) ? ~u : (u | 0x80000000u);
}
__device__ __forceinline__ float dec_f32(unsigned s) {
    unsigned u = (s & 0x80000000u) ? (s & 0x7FFFFFFFu) : ~s;
    return __uint_as_float(u);
}
// enc(-inf) = ~0xFF800000 = 0x007FFFFF
#define ENC_NEG_INF 0x007FFFFFu

// ---------------------------------------------------------------------------
// Kernel 1: row-wise L2 normalization (f32 in) -> fp16 out.  One block/row.
// ---------------------------------------------------------------------------
__global__ __launch_bounds__(256) void rownorm_kernel(const float* __restrict__ src,
                                                      _Float16* __restrict__ dst) {
    const int row = blockIdx.x;
    const float* r = src + (size_t)row * D_DIM;
    _Float16*    w = dst + (size_t)row * D_DIM;

    float ss = 0.f;
    for (int i = threadIdx.x; i < D_DIM; i += 256) {
        float t = r[i];
        ss += t * t;
    }
    #pragma unroll
    for (int off = 16; off > 0; off >>= 1) ss += __shfl_xor(ss, off, 32);

    __shared__ float part[8];
    __shared__ float scale_s;
    if ((threadIdx.x & 31) == 0) part[threadIdx.x >> 5] = ss;
    __syncthreads();
    if (threadIdx.x == 0) {
        float t = 0.f;
        #pragma unroll
        for (int i = 0; i < 8; ++i) t += part[i];
        scale_s = 1.0f / fmaxf(sqrtf(t), 1e-8f);
    }
    __syncthreads();
    const float scale = scale_s;
    for (int i = threadIdx.x; i < D_DIM; i += 256) {
        w[i] = (_Float16)(r[i] * scale);
    }
}

// ---------------------------------------------------------------------------
// Kernel 2: init row/col maxima to encoded -inf.
// ---------------------------------------------------------------------------
__global__ void initmax_kernel(unsigned* __restrict__ p, int n) {
    int i = blockIdx.x * 256 + threadIdx.x;
    if (i < n) p[i] = ENC_NEG_INF;
}

// ---------------------------------------------------------------------------
// Kernel 3: WMMA GEMM  C = Xn * Yn^T  (row-major fp16, K = 1024), fused with
// row-max / col-max reduction.
//
// Block = 256 threads = 8 waves covering a 128x128 C tile.
// Each wave computes a 32x64 block: 2 A fragments x 4 B fragments -> 8 WMMAs
// per 32-wide K-step with each B fragment reused by two WMMAs
// (12 b128 loads per 8 WMMAs instead of 18).
// ---------------------------------------------------------------------------
union AFrag { v16h v; v8h h[2]; };

__global__ __launch_bounds__(256) void gemm_max_kernel(const _Float16* __restrict__ X,
                                                       const _Float16* __restrict__ Y,
                                                       unsigned* __restrict__ rowmax,
                                                       unsigned* __restrict__ colmax) {
    const int lane    = threadIdx.x & 31;
    const int wave    = threadIdx.x >> 5;
    const int l16     = lane & 15;
    const int halfsel = lane >> 4;          // 0 for lanes 0-15, 1 for 16-31
    const int mstrip  = wave >> 1;          // 0..3 : which 32-row strip
    const int nhalf   = wave & 1;           // 0..1 : which 64-col half
    const int row0    = blockIdx.y * 128;
    const int col0    = blockIdx.x * 128;
    const int mrow    = row0 + mstrip * 32;
    const int ncol    = col0 + nhalf * 64;

    v8f acc[2][4] = {};

    // A fragments: lane l16 holds rows (mrow+l16) and (mrow+16+l16);
    // lanes 0-15 hold K = base+0..15,
    // lanes 16-31 hold K = base+16..31 (same order as B).
    const _Float16* xrow0 = X + (size_t)(mrow + l16) * D_DIM;
    const _Float16* xrow1 = X + (size_t)(mrow + 16 + l16) * D_DIM;
    const int akoff = halfsel * 16;

    // B fragments: lane l16 holds column (ncol + j*16 + l16) of C, i.e. that
    // row of Y; lanes 0-15 hold K base+0..15, lanes 16-31 hold K base+16..31.
    const _Float16* yrow[4];
    #pragma unroll
    for (int j = 0; j < 4; ++j)
        yrow[j] = Y + (size_t)(ncol + j * 16 + l16) * D_DIM + halfsel * 16;

    for (int kk = 0; kk < D_DIM; kk += 32) {
        AFrag a0, a1;
        a0.h[0] = *(const v8h*)(xrow0 + kk + akoff);
        a0.h[1] = *(const v8h*)(xrow0 + kk + akoff + 8);
        a1.h[0] = *(const v8h*)(xrow1 + kk + akoff);
        a1.h[1] = *(const v8h*)(xrow1 + kk + akoff + 8);
        if (kk + 64 <= D_DIM - 32) {
            __builtin_prefetch(xrow0 + kk + 64, 0, 1);   // global_prefetch_b8
            __builtin_prefetch(xrow1 + kk + 64, 0, 1);
        }
        #pragma unroll
        for (int j = 0; j < 4; ++j) {
            v16h b = *(const v16h*)(yrow[j] + kk);
            acc[0][j] = __builtin_amdgcn_wmma_f32_16x16x32_f16(
                false, a0.v, false, b, (short)0, acc[0][j], false, false);
            acc[1][j] = __builtin_amdgcn_wmma_f32_16x16x32_f16(
                false, a1.v, false, b, (short)0, acc[1][j], false, false);
        }
    }

    // C/D layout: VGPR r, lanes 0-15 -> M=r, lanes 16-31 -> M=r+8; N = lane&15.
    __shared__ float rowbuf[2][128];   // [nhalf][local row]
    __shared__ float colbuf[4][128];   // [mstrip][local col]

    // ---- per-wave row maxima (max over this wave's 64 columns) ----
    #pragma unroll
    for (int i = 0; i < 2; ++i) {
        #pragma unroll
        for (int r = 0; r < 8; ++r) {
            float v = acc[i][0][r];
            #pragma unroll
            for (int j = 1; j < 4; ++j) v = fmaxf(v, acc[i][j][r]);
            v = fmaxf(v, __shfl_xor(v, 1, 32));
            v = fmaxf(v, __shfl_xor(v, 2, 32));
            v = fmaxf(v, __shfl_xor(v, 4, 32));
            v = fmaxf(v, __shfl_xor(v, 8, 32));
            if (l16 == 0)
                rowbuf[nhalf][mstrip * 32 + i * 16 + halfsel * 8 + r] = v;
        }
    }

    // ---- per-wave col maxima (max over this wave's 32 rows) ----
    #pragma unroll
    for (int j = 0; j < 4; ++j) {
        float v = acc[0][j][0];
        #pragma unroll
        for (int r = 1; r < 8; ++r) v = fmaxf(v, acc[0][j][r]);
        #pragma unroll
        for (int r = 0; r < 8; ++r) v = fmaxf(v, acc[1][j][r]);
        v = fmaxf(v, __shfl_xor(v, 16, 32));
        if (halfsel == 0)
            colbuf[mstrip][nhalf * 64 + j * 16 + l16] = v;
    }
    __syncthreads();

    // ---- combine waves, one encoded atomicMax per row/col of the tile ----
    if (threadIdx.x < 128) {
        const int t = threadIdx.x;
        float rv = fmaxf(rowbuf[0][t], rowbuf[1][t]);
        atomicMax(&rowmax[row0 + t], enc_f32(rv));

        float cv = colbuf[0][t];
        #pragma unroll
        for (int s = 1; s < 4; ++s) cv = fmaxf(cv, colbuf[s][t]);
        atomicMax(&colmax[col0 + t], enc_f32(cv));
    }
}

// ---------------------------------------------------------------------------
// Kernel 4: log-prob + sum of both maxima vectors. One block.
// logp(x) = -0.5*((x-1)/0.3)^2 - (log(0.3) + 0.5*log(2*pi))
// ---------------------------------------------------------------------------
__global__ __launch_bounds__(256) void finalize_kernel(const unsigned* __restrict__ rowmax,
                                                       const unsigned* __restrict__ colmax,
                                                       float* __restrict__ out, int n, int m) {
    const float INV_SIGMA = 1.0f / 0.3f;
    const float NEG_LOG_NORM = 0.28503427f;   // -(log(0.3)+0.5*log(2pi))

    float s1 = 0.f, s2 = 0.f;
    for (int i = threadIdx.x; i < n; i += 256) {
        float z = (dec_f32(rowmax[i]) - 1.0f) * INV_SIGMA;
        s1 += -0.5f * z * z + NEG_LOG_NORM;
    }
    for (int i = threadIdx.x; i < m; i += 256) {
        float z = (dec_f32(colmax[i]) - 1.0f) * INV_SIGMA;
        s2 += -0.5f * z * z + NEG_LOG_NORM;
    }
    #pragma unroll
    for (int off = 16; off > 0; off >>= 1) {
        s1 += __shfl_xor(s1, off, 32);
        s2 += __shfl_xor(s2, off, 32);
    }
    __shared__ float r1[8], r2[8];
    if ((threadIdx.x & 31) == 0) {
        r1[threadIdx.x >> 5] = s1;
        r2[threadIdx.x >> 5] = s2;
    }
    __syncthreads();
    if (threadIdx.x == 0) {
        float t1 = 0.f, t2 = 0.f;
        #pragma unroll
        for (int w = 0; w < 8; ++w) { t1 += r1[w]; t2 += r2[w]; }
        out[0] = t1;
        out[1] = t2;
    }
}

// ---------------------------------------------------------------------------
extern "C" void kernel_launch(void* const* d_in, const int* in_sizes, int n_in,
                              void* d_out, int out_size, void* d_ws, size_t ws_size,
                              hipStream_t stream) {
    const float* ex = (const float*)d_in[0];
    const float* ey = (const float*)d_in[1];
    float* out = (float*)d_out;

    const int N = in_sizes[0] / D_DIM;   // 8192
    const int M = in_sizes[1] / D_DIM;   // 8192

    char* ws = (char*)d_ws;
    _Float16* Xh     = (_Float16*)ws;
    _Float16* Yh     = (_Float16*)(ws + (size_t)N * D_DIM * sizeof(_Float16));
    unsigned* rowmax = (unsigned*)(ws + (size_t)(N + M) * D_DIM * sizeof(_Float16));
    unsigned* colmax = rowmax + N;

    // 1) normalize + fp16 cast
    rownorm_kernel<<<dim3(N), 256, 0, stream>>>(ex, Xh);
    rownorm_kernel<<<dim3(M), 256, 0, stream>>>(ey, Yh);

    // 2) init maxima
    initmax_kernel<<<dim3((N + M + 255) / 256), 256, 0, stream>>>(rowmax, N + M);

    // 3) WMMA GEMM + fused max reduction (128x128 C tile per workgroup)
    gemm_max_kernel<<<dim3(M / 128, N / 128), 256, 0, stream>>>(Xh, Yh, rowmax, colmax);

    // 4) log-prob sums
    finalize_kernel<<<dim3(1), 256, 0, stream>>>(rowmax, colmax, out, N, M);
}
